// AttentionBlock_10969346474415
// MI455X (gfx1250) — matrix operand order repaired and checked by reading, not compiled, not run on hardware
//
#include <hip/hip_runtime.h>
#include <hip/hip_bf16.h>

typedef _Float16     v16h __attribute__((ext_vector_type(16)));
typedef float        v8f  __attribute__((ext_vector_type(8)));
typedef unsigned int v4u  __attribute__((ext_vector_type(4)));

namespace {
constexpr int BN = 16;    // batch
constexpr int C  = 256;   // channels
constexpr int N  = 1024;  // H*W
constexpr int G  = 8;     // groups
constexpr int CG = C / G; // 32 channels per group
constexpr float SCALE = 0.0625f; // C^-0.5
}

union HVec { v4u u[2]; v16h h; };

static __device__ __forceinline__ unsigned short f2h(float f) {
    _Float16 h = (_Float16)f;
    return __builtin_bit_cast(unsigned short, h);
}

static __device__ __forceinline__ v8f wmma16(v16h a, v16h b, v8f c) {
    return __builtin_amdgcn_wmma_f32_16x16x32_f16(false, a, false, b, (short)0, c, false, false);
}

// Async DMA: copy 64B per lane (4 x 16B), global -> LDS, no VGPR round-trip (ASYNCcnt).
// IOFFSET is added to BOTH the LDS and global addresses (ISA 10.x async ops).
static __device__ __forceinline__ void async_cp64(unsigned lds, const void* gaddr) {
    asm volatile("global_load_async_to_lds_b128 %0, %1, off offset:0"  :: "v"(lds), "v"(gaddr) : "memory");
    asm volatile("global_load_async_to_lds_b128 %0, %1, off offset:16" :: "v"(lds), "v"(gaddr) : "memory");
    asm volatile("global_load_async_to_lds_b128 %0, %1, off offset:32" :: "v"(lds), "v"(gaddr) : "memory");
    asm volatile("global_load_async_to_lds_b128 %0, %1, off offset:48" :: "v"(lds), "v"(gaddr) : "memory");
}

// ---------------------------------------------------------------------------
// 1) GroupNorm: x[B,C,N] f32 -> hb[B,N,C] f16  (row-major over C: A-operand layout)
// ---------------------------------------------------------------------------
__global__ __launch_bounds__(256) void groupnorm_k(const float* __restrict__ x,
                                                   const float* __restrict__ nw,
                                                   const float* __restrict__ nb,
                                                   unsigned short* __restrict__ hb) {
    const int b = blockIdx.x >> 3;
    const int g = blockIdx.x & 7;
    const int tid = threadIdx.x;
    const float* xg = x + ((size_t)b * C + (size_t)g * CG) * N;

    float s = 0.f, s2 = 0.f;
    for (int i = tid; i < CG * N; i += 256) {
        float v = xg[i];
        s += v; s2 += v * v;
    }
    __shared__ float rs[256], rq[256];
    rs[tid] = s; rq[tid] = s2;
    __syncthreads();
    for (int off = 128; off > 0; off >>= 1) {
        if (tid < off) { rs[tid] += rs[tid + off]; rq[tid] += rq[tid + off]; }
        __syncthreads();
    }
    const float inv = 1.f / (float)(CG * N);
    const float mean = rs[0] * inv;
    const float var  = rq[0] * inv - mean * mean;
    const float rstd = rsqrtf(var + 1e-5f);

    for (int i = tid; i < CG * N; i += 256) {
        const int cl = i & (CG - 1);
        const int n  = i >> 5;
        const int c  = g * CG + cl;
        const float v = x[((size_t)b * C + c) * N + n];
        const float hn = (v - mean) * rstd * nw[c] + nb[c];
        hb[((size_t)b * N + n) * C + c] = f2h(hn);
    }
}

// ---------------------------------------------------------------------------
// 2) Weight convert f32 -> f16 (row-major [O,C] kept: B-operand layout)
// ---------------------------------------------------------------------------
__global__ __launch_bounds__(256) void convert_w_k(const float* __restrict__ qkvw,
                                                   const float* __restrict__ projw,
                                                   unsigned short* __restrict__ wq,
                                                   unsigned short* __restrict__ wp) {
    const int i = blockIdx.x * 256 + threadIdx.x;
    if (i < 3 * C * C) wq[i] = f2h(qkvw[i]);
    if (i < C * C)     wp[i] = f2h(projw[i]);
}

// ---------------------------------------------------------------------------
// 3) QKV: qkv[n,o] = sum_c hb[n,c] * w[o,c] + bias. One 16x16 tile per wave.
//    q -> qb[B,N,C] (pre-scaled), k -> kb[B,N,C], v -> vtb[B,C,N] (transposed)
// ---------------------------------------------------------------------------
__global__ __launch_bounds__(256) void qkv_k(const unsigned short* __restrict__ hb,
                                             const unsigned short* __restrict__ wq,
                                             const float* __restrict__ qkvb,
                                             unsigned short* __restrict__ qb,
                                             unsigned short* __restrict__ kb,
                                             unsigned short* __restrict__ vtb) {
    const int wave = threadIdx.x >> 5, lane = threadIdx.x & 31;
    const int m = lane & 15, hi = lane >> 4;
    const int n0 = blockIdx.x * 128 + wave * 16;
    const int o0 = blockIdx.y * 16;
    const int b  = blockIdx.z;

    const unsigned short* arow = hb + ((size_t)b * N + n0 + m) * C + hi * 16;
    const unsigned short* brow = wq + ((size_t)(o0 + m)) * C + hi * 16;

    v8f acc = {};
#pragma unroll
    for (int cc = 0; cc < 8; cc++) {
        HVec A, B;
        A.u[0] = *(const v4u*)(arow + cc * 32);
        A.u[1] = *(const v4u*)(arow + cc * 32 + 8);
        B.u[0] = *(const v4u*)(brow + cc * 32);
        B.u[1] = *(const v4u*)(brow + cc * 32 + 8);
        acc = wmma16(A.h, B.h, acc);
    }

    const int o = o0 + m;
    const float bias = qkvb[o];
#pragma unroll
    for (int r = 0; r < 8; r++) {
        const int n = n0 + r + hi * 8;
        const float v = acc[r] + bias;
        if (o0 < 256) {
            qb[((size_t)b * N + n) * C + o] = f2h(v * SCALE);
        } else if (o0 < 512) {
            kb[((size_t)b * N + n) * C + (o - 256)] = f2h(v);
        } else {
            vtb[((size_t)b * C + (o - 512)) * N + n] = f2h(v);
        }
    }
}

// ---------------------------------------------------------------------------
// 4) Flash attention. Block = 8 waves x 16 rows = 128 query rows of one batch.
//    Double-buffered LDS K/V chunks filled by async global->LDS DMA so the
//    next chunk's 32KB streams in behind the current chunk's 32 WMMAs.
// ---------------------------------------------------------------------------
__global__ __launch_bounds__(256) void attn_k(const unsigned short* __restrict__ qb,
                                              const unsigned short* __restrict__ kb,
                                              const unsigned short* __restrict__ vtb,
                                              unsigned short* __restrict__ ob) {
    __shared__ unsigned short kt[2][32 * 256];   // K chunk rows [32][C], x2 buffers
    __shared__ unsigned short vt[2][256 * 32];   // V^T rows [C][32],    x2 buffers
    __shared__ unsigned short pb[8 * 16 * 32];   // per-wave P staging

    const int tid = threadIdx.x;
    const int wave = tid >> 5, lane = tid & 31;
    const int m = lane & 15, hi = lane >> 4;
    const int b  = blockIdx.y;
    const int i0 = blockIdx.x * 128 + wave * 16;

    // Persistent Q A-fragments for the wave's 16 rows (8 chunks of K=32 over C)
    HVec Aq[8];
    const unsigned short* qrow = qb + ((size_t)b * N + i0 + m) * C + hi * 16;
#pragma unroll
    for (int cc = 0; cc < 8; cc++) {
        Aq[cc].u[0] = *(const v4u*)(qrow + cc * 32);
        Aq[cc].u[1] = *(const v4u*)(qrow + cc * 32 + 8);
    }

    float mrow[8], lrow[8];
#pragma unroll
    for (int r = 0; r < 8; r++) { mrow[r] = -1e30f; lrow[r] = 0.f; }
    v8f Oacc[16];
#pragma unroll
    for (int t = 0; t < 16; t++) Oacc[t] = (v8f){};

    unsigned short* pw = pb + wave * 512;

    // Per-thread async copy slices (64B of K rows + 64B of V^T rows per chunk)
    const int jr = tid >> 3, cs = (tid & 7) << 5;
    const unsigned short* gk_base = kb + ((size_t)b * N + jr) * C + cs;
    const unsigned short* gv_base = vtb + ((size_t)b * C + tid) * N;

#define PREFETCH_CHUNK(JC, BUF)                                                    \
    do {                                                                           \
        async_cp64((unsigned)(size_t)&kt[(BUF)][jr * 256 + cs],                    \
                   gk_base + (size_t)(JC) * 32 * C);                               \
        async_cp64((unsigned)(size_t)&vt[(BUF)][tid * 32],                         \
                   gv_base + (JC) * 32);                                           \
    } while (0)

    PREFETCH_CHUNK(0, 0);

    for (int jc = 0; jc < N / 32; jc++) {
        const int cur = jc & 1;
        // my lanes' async copies for chunk jc have landed
        asm volatile("s_wait_asynccnt 0" ::: "memory");
        // all waves' copies landed; all waves done reading buffer cur^1
        __syncthreads();
        if (jc + 1 < N / 32) PREFETCH_CHUNK(jc + 1, cur ^ 1);

        // S = Q K^T for 2 sub-tiles of 16 cols (full 32-col chunk)
        v8f S0 = {}, S1 = {};
#pragma unroll
        for (int cc = 0; cc < 8; cc++) {
            HVec B0, B1;
            const unsigned short* k0 = &kt[cur][m * 256 + cc * 32 + hi * 16];
            const unsigned short* k1 = &kt[cur][(16 + m) * 256 + cc * 32 + hi * 16];
            B0.u[0] = *(const v4u*)(k0); B0.u[1] = *(const v4u*)(k0 + 8);
            B1.u[0] = *(const v4u*)(k1); B1.u[1] = *(const v4u*)(k1 + 8);
            S0 = wmma16(Aq[cc].h, B0.h, S0);
            S1 = wmma16(Aq[cc].h, B1.h, S1);
        }

        // Online softmax update; P written to LDS in A-operand row-major layout
        float corr[8];
#pragma unroll
        for (int r = 0; r < 8; r++) {
            float mx = fmaxf(S0[r], S1[r]);
            mx = fmaxf(mx, __shfl_xor(mx, 1, 32));
            mx = fmaxf(mx, __shfl_xor(mx, 2, 32));
            mx = fmaxf(mx, __shfl_xor(mx, 4, 32));
            mx = fmaxf(mx, __shfl_xor(mx, 8, 32));
            const float mnew = fmaxf(mrow[r], mx);
            corr[r] = __expf(mrow[r] - mnew);
            mrow[r] = mnew;
            const float p0 = __expf(S0[r] - mnew);
            const float p1 = __expf(S1[r] - mnew);
            float rsum = p0 + p1;
            rsum += __shfl_xor(rsum, 1, 32);
            rsum += __shfl_xor(rsum, 2, 32);
            rsum += __shfl_xor(rsum, 4, 32);
            rsum += __shfl_xor(rsum, 8, 32);
            lrow[r] = lrow[r] * corr[r] + rsum;
            const int M = r + hi * 8;
            pw[M * 32 + m]      = f2h(p0);
            pw[M * 32 + 16 + m] = f2h(p1);
        }
#pragma unroll
        for (int t = 0; t < 16; t++) {
#pragma unroll
            for (int r = 0; r < 8; r++) Oacc[t][r] *= corr[r];
        }

        // make per-wave P staging visible across lanes (LDS in-order per wave)
        asm volatile("s_wait_dscnt 0" ::: "memory");

        HVec Pa;
        const unsigned short* pr = pw + m * 32 + hi * 16;
        Pa.u[0] = *(const v4u*)(pr);
        Pa.u[1] = *(const v4u*)(pr + 8);

        // O += P (16x32) x V (32x16) for each of the 16 output-channel tiles
#pragma unroll
        for (int t = 0; t < 16; t++) {
            HVec Bv;
            const unsigned short* vr = &vt[cur][(t * 16 + m) * 32 + hi * 16];
            Bv.u[0] = *(const v4u*)(vr);
            Bv.u[1] = *(const v4u*)(vr + 8);
            Oacc[t] = wmma16(Pa.h, Bv.h, Oacc[t]);
        }
    }
#undef PREFETCH_CHUNK

#pragma unroll
    for (int r = 0; r < 8; r++) lrow[r] = 1.f / lrow[r];
#pragma unroll
    for (int t = 0; t < 16; t++) {
#pragma unroll
        for (int r = 0; r < 8; r++) {
            const int n = i0 + r + hi * 8;
            ob[((size_t)b * N + n) * C + t * 16 + m] = f2h(Oacc[t][r] * lrow[r]);
        }
    }
}

// ---------------------------------------------------------------------------
// 5) Proj + bias + residual: out[b,o,n] = x[b,o,n] + sum_c ob[n,c]*w[o,c] + pb[o]
// ---------------------------------------------------------------------------
__global__ __launch_bounds__(256) void proj_k(const unsigned short* __restrict__ ob,
                                              const unsigned short* __restrict__ wp,
                                              const float* __restrict__ projb,
                                              const float* __restrict__ x,
                                              float* __restrict__ out) {
    const int wave = threadIdx.x >> 5, lane = threadIdx.x & 31;
    const int m = lane & 15, hi = lane >> 4;
    const int n0 = blockIdx.x * 128 + wave * 16;
    const int o0 = blockIdx.y * 16;
    const int b  = blockIdx.z;

    const unsigned short* arow = ob + ((size_t)b * N + n0 + m) * C + hi * 16;
    const unsigned short* brow = wp + ((size_t)(o0 + m)) * C + hi * 16;

    v8f acc = {};
#pragma unroll
    for (int cc = 0; cc < 8; cc++) {
        HVec A, B;
        A.u[0] = *(const v4u*)(arow + cc * 32);
        A.u[1] = *(const v4u*)(arow + cc * 32 + 8);
        B.u[0] = *(const v4u*)(brow + cc * 32);
        B.u[1] = *(const v4u*)(brow + cc * 32 + 8);
        acc = wmma16(A.h, B.h, acc);
    }

    const int o = o0 + m;
    const float bias = projb[o];
#pragma unroll
    for (int r = 0; r < 8; r++) {
        const int n = n0 + r + hi * 8;
        const size_t oi = ((size_t)b * C + o) * N + n;
        out[oi] = x[oi] + acc[r] + bias;
    }
}

// ---------------------------------------------------------------------------
extern "C" void kernel_launch(void* const* d_in, const int* in_sizes, int n_in,
                              void* d_out, int out_size, void* d_ws, size_t ws_size,
                              hipStream_t stream) {
    const float* x     = (const float*)d_in[0];
    const float* nw    = (const float*)d_in[1];
    const float* nb    = (const float*)d_in[2];
    const float* qkvw  = (const float*)d_in[3];
    const float* qkvb  = (const float*)d_in[4];
    const float* projw = (const float*)d_in[5];
    const float* projb = (const float*)d_in[6];
    float* out = (float*)d_out;

    char* base = (char*)d_ws;
    const size_t HB = (size_t)BN * N * C * sizeof(unsigned short); // 8 MB
    unsigned short* hb  = (unsigned short*)(base + 0 * HB);
    unsigned short* qb  = (unsigned short*)(base + 1 * HB);
    unsigned short* kb  = (unsigned short*)(base + 2 * HB);
    unsigned short* vtb = (unsigned short*)(base + 3 * HB);
    unsigned short* ob  = (unsigned short*)(base + 4 * HB);
    unsigned short* wq  = (unsigned short*)(base + 5 * HB);
    unsigned short* wp  = (unsigned short*)(base + 5 * HB + (size_t)3 * C * C * sizeof(unsigned short));

    groupnorm_k<<<BN * G, 256, 0, stream>>>(x, nw, nb, hb);
    convert_w_k<<<(3 * C * C) / 256, 256, 0, stream>>>(qkvw, projw, wq, wp);
    qkv_k<<<dim3(N / 128, (3 * C) / 16, BN), 256, 0, stream>>>(hb, wq, qkvb, qb, kb, vtb);
    attn_k<<<dim3(N / 128, BN), 256, 0, stream>>>(qb, kb, vtb, ob);
    proj_k<<<dim3(N / 128, C / 16, BN), 256, 0, stream>>>(ob, wp, projb, x, out);
}
